// MultiHeadAttention_35184372089471
// MI455X (gfx1250) — matrix operand order repaired and checked by reading, not compiled, not run on hardware
//
#include <hip/hip_runtime.h>

// ---------------------------------------------------------------------------
// Types / fragments for CDNA5 WMMA (wave32): v_wmma_f32_16x16x32_bf16
// ---------------------------------------------------------------------------
typedef __bf16 bf16x16 __attribute__((ext_vector_type(16)));
typedef float  f32x8   __attribute__((ext_vector_type(8)));
typedef float  f32x4   __attribute__((ext_vector_type(4)));
typedef unsigned short u16x4 __attribute__((ext_vector_type(4)));
typedef unsigned int v4u __attribute__((ext_vector_type(4)));
typedef int          v4i __attribute__((ext_vector_type(4)));
typedef int          v8i __attribute__((ext_vector_type(8)));

union FragB16 {
    bf16x16        v;
    unsigned short us[16];
    uint4          q[2];
};

__device__ __forceinline__ f32x8 wmma_bf16(bf16x16 a, bf16x16 b, f32x8 c) {
    // (neg_a, A, neg_b, B, c_mod, C, reuse_a, reuse_b)
    return __builtin_amdgcn_wmma_f32_16x16x32_bf16(false, a, false, b,
                                                   (short)0, c, false, false);
}

__device__ __forceinline__ unsigned short f2bf(float f) {
    unsigned int u = __float_as_uint(f);
    u += 0x7FFFu + ((u >> 16) & 1u);            // round-to-nearest-even
    return (unsigned short)(u >> 16);
}

// ds_swizzle XOR within 32 lanes: imm = (xor_mask<<10) | and_mask(0x1F)
template <int IMM>
__device__ __forceinline__ float swz_xor(float x) {
    return __int_as_float(__builtin_amdgcn_ds_swizzle(__float_as_int(x), IMM));
}

// Problem constants
constexpr int Bsz = 8, Ssz = 16, Dch = 256, Hh = 8, DP = 32;
constexpr int FRAMES = Bsz * Ssz;               // 128
constexpr int KTOT   = Dch * 9;                 // 2304  (k = tap*256 + i)
constexpr size_t AELEM = (size_t)FRAMES * Dch * 1024;       // 33,554,432
constexpr size_t WELEM = (size_t)KTOT * Dch;                // 589,824

// ---------------------------------------------------------------------------
// f32 -> bf16 bulk convert (8 elems / thread), read-once -> nontemporal loads
// ---------------------------------------------------------------------------
__global__ __launch_bounds__(256) void cvt_bf16(const float* __restrict__ src,
                                                unsigned short* __restrict__ dst) {
    const size_t i = ((size_t)blockIdx.x * 256 + threadIdx.x) * 8;
    f32x4 a = __builtin_nontemporal_load((const f32x4*)(src + i));
    f32x4 b = __builtin_nontemporal_load((const f32x4*)(src + i + 4));
    u16x4 lo = { f2bf(a.x), f2bf(a.y), f2bf(a.z), f2bf(a.w) };
    u16x4 hi = { f2bf(b.x), f2bf(b.y), f2bf(b.z), f2bf(b.w) };
    *(u16x4*)(dst + i)     = lo;
    *(u16x4*)(dst + i + 4) = hi;
}

// ---------------------------------------------------------------------------
// Pack conv weights (O,I,3,3) f32 -> bf16 [o][k], k = (fy*3+fx)*256 + i
// ---------------------------------------------------------------------------
__global__ void pack_weights(const float* __restrict__ w,
                             unsigned short* __restrict__ W2t) {
    int idx = blockIdx.x * 256 + threadIdx.x;
    if (idx >= (int)WELEM) return;
    int o   = idx / KTOT;
    int k   = idx - o * KTOT;
    int tap = k >> 8;
    int i   = k & 255;
    int fy  = tap / 3, fx = tap - fy * 3;
    W2t[idx] = f2bf(w[(((size_t)o * Dch + i) * 3 + fy) * 3 + fx]);
}

// ---------------------------------------------------------------------------
// TDM: load a 3-D weight slab into LDS:
//   x = 32 contiguous channels (at i0), y = 9 taps (stride 256),
//   z = 64 o-rows (stride 2304)  ->  LDS layout [o][tap][i], 36,864 B.
// ---------------------------------------------------------------------------
__device__ __forceinline__ void tdm_load_wslab(const unsigned short* W2,
                                               unsigned lds_addr,
                                               int o0, int i0) {
    const unsigned long long ga =
        (unsigned long long)(uintptr_t)W2 +
        ((unsigned long long)o0 * KTOT + (unsigned long long)i0) * 2ull;
    // D# group0: count=1 | lds_addr | global_addr[56:0] | type=2
    v4u g0 = { 1u, lds_addr, (unsigned)ga,
               (unsigned)((ga >> 32) & 0x01FFFFFFull) | 0x80000000u };
    // group1: data_size=2B; tensor_dim0=2304; tensor_dim1=9;
    //         tile_dim0=32; tile_dim1=9; tile_dim2=64;
    //         dim0_stride=256 (tap step); dim1_stride=2304 (o step)
    v8i g1 = { 0x00010000,        // d0: data_size=1(2B) <<16
               0x09000000,        // d1[31:16] = tensor_dim0 lo16 = 2304
               0x00090000,        // d2[31:16] = tensor_dim1 lo16 = 9
               0x00200000,        // d3[31:16] = tile_dim0 = 32
               (64 << 16) | 9,    // d4: tile_dim1=9, tile_dim2=64
               256,               // d5: dim0_stride lo32 = 256
               0x09000000,        // d6[31:16] = dim1_stride lo16 = 2304
               0 };               // d7: dim1_stride hi = 0
    // group2: tensor_dim2 = 256 (o rows); 3rd-dim stride/tile_dim3 unused
    v4i g2 = { 256, 0, 0, 0 };
    v4i g3 = { 0, 0, 0, 0 };
#if defined(__clang_major__) && __clang_major__ >= 23
    v8i gz8 = { 0, 0, 0, 0, 0, 0, 0, 0 };
    __builtin_amdgcn_tensor_load_to_lds(g0, g1, g2, g3, gz8, 0);
#else
    __builtin_amdgcn_tensor_load_to_lds(g0, g1, g2, g3, 0);
#endif
}

// ---------------------------------------------------------------------------
// Implicit-GEMM 3x3 SAME conv via WMMA bf16 (input already bf16).
//   Block: 8 waves -> 256 positions (8 rows) x 64 out-channels.
//   Channel-group-major: per 32-channel group, stage one 10x34 halo tile in
//   LDS and one [64][9][32] weight slab (TDM), then sweep all 9 taps.
//   Wave tile: 64(o) x 32(pos) = 8 f32x8 accumulators, 72 WMMA / group.
// ---------------------------------------------------------------------------
template <bool OUT_F32>
__global__ __launch_bounds__(256) void conv3x3_wmma(
    const unsigned short* __restrict__ xin,  // bf16 (128,256,32,32)
    const unsigned short* __restrict__ W2,   // bf16 [o][KTOT]
    const float* __restrict__ bias,
    void* __restrict__ yout)
{
    __shared__ unsigned short Wsl[64 * 9 * 32];   // [o][tap][i]   36,864 B
    __shared__ unsigned short Xh[10 * 34 * 32];   // [row][col][i] 21,760 B

    const int frame = blockIdx.y;
    const int mb    = blockIdx.x & 3;         // 4 position blocks of 256
    const int nb    = blockIdx.x >> 2;        // 4 channel blocks of 64
    const int r0    = mb * 8;
    const int o0    = nb * 64;
    const int tid   = threadIdx.x;
    const int lane  = tid & 31;
    const int wave  = tid >> 5;
    const int half  = lane >> 4;
    const int li    = lane & 15;

    f32x8 acc[4][2];
#pragma unroll
    for (int i = 0; i < 4; ++i)
#pragma unroll
        for (int j = 0; j < 2; ++j) acc[i][j] = (f32x8){0, 0, 0, 0, 0, 0, 0, 0};

    const unsigned ws_lds = (unsigned)(uintptr_t)(&Wsl[0]);

    for (int ig = 0; ig < 8; ++ig) {
        const int i0 = ig * 32;

        // --- weight slab via Tensor Data Mover (one wave issues)
        if (wave == 0) {
            tdm_load_wslab(W2, ws_lds, o0, i0);
            __builtin_amdgcn_s_wait_tensorcnt((short)0);
        }
        // --- halo tile: rows r0-1..r0+8, cols -1..32, 32 channels
        for (int cell = tid; cell < 340; cell += 256) {
            const int row = cell / 34;            // 0..9
            const int col = cell - row * 34;      // 0..33
            const int r   = r0 + row - 1;
            const int c   = col - 1;
            unsigned short* dst = Xh + cell * 32;
            if (((unsigned)r < 32u) && ((unsigned)c < 32u)) {
                const size_t bx =
                    ((size_t)frame * Dch + i0) * 1024 + (size_t)r * 32 + c;
#pragma unroll
                for (int j = 0; j < 32; ++j) dst[j] = xin[bx + (size_t)j * 1024];
            } else {
#pragma unroll
                for (int j = 0; j < 32; ++j) dst[j] = 0;
            }
        }
        __syncthreads();

        // --- sweep 9 taps out of the halo tile
#pragma unroll
        for (int tap = 0; tap < 9; ++tap) {
            const int fy = tap / 3;
            const int fx = tap - fy * 3;
            // B fragments (activations 32x16): lane = position, K contig 16
            FragB16 bfrag[2];
#pragma unroll
            for (int pt = 0; pt < 2; ++pt) {
                const int cell = (wave + fy) * 34 + (pt * 16 + li + fx);
                const unsigned short* rowp = Xh + cell * 32 + half * 16;
                bfrag[pt].q[0] = *(const uint4*)(rowp);
                bfrag[pt].q[1] = *(const uint4*)(rowp + 8);
            }
            // A fragments (weights 16x32): lane = o, K contig 16
#pragma unroll
            for (int ot = 0; ot < 4; ++ot) {
                FragB16 afrag;
                const unsigned short* rowp =
                    Wsl + ((ot * 16 + li) * 9 + tap) * 32 + half * 16;
                afrag.q[0] = *(const uint4*)(rowp);
                afrag.q[1] = *(const uint4*)(rowp + 8);
                acc[ot][0] = wmma_bf16(afrag.v, bfrag[0].v, acc[ot][0]);
                acc[ot][1] = wmma_bf16(afrag.v, bfrag[1].v, acc[ot][1]);
            }
        }
        __syncthreads();
    }

    // --- epilogue: bias + store. D layout: VGPR v -> M(=o), lane&15 -> N(=pos)
#pragma unroll
    for (int ot = 0; ot < 4; ++ot) {
#pragma unroll
        for (int pt = 0; pt < 2; ++pt) {
            const int p = mb * 256 + wave * 32 + pt * 16 + li;
#pragma unroll
            for (int v = 0; v < 8; ++v) {
                const int o = o0 + ot * 16 + half * 8 + v;
                const float val = acc[ot][pt][v] + bias[o];
                const size_t a  = ((size_t)frame * Dch + o) * 1024 + p;
                if constexpr (OUT_F32)
                    __builtin_nontemporal_store(val, (float*)yout + a);
                else
                    ((unsigned short*)yout)[a] = f2bf(val);
            }
        }
    }
}

// ---------------------------------------------------------------------------
// Attention: one wave per (b,h,s,r) group.  scores = Q K^T / sqrt(32),
// softmax over cols, out = attn V.  qp/kp/vp bf16 frame-major (128,256,32,32).
// ---------------------------------------------------------------------------
__global__ __launch_bounds__(256) void attn_wmma(
    const unsigned short* __restrict__ qp,
    const unsigned short* __restrict__ kp,
    const unsigned short* __restrict__ vp,
    float* __restrict__ attn_out,            // (B,H,S,R,32,32) f32
    unsigned short* __restrict__ concat)     // (128,256,32,32) bf16
{
    __shared__ unsigned short As[8][32 * 32]; // per-wave attn tile, bf16

    const int tid  = threadIdx.x;
    const int lane = tid & 31;
    const int wave = tid >> 5;
    const int half = lane >> 4;
    const int li   = lane & 15;

    const int g = blockIdx.x * 8 + wave;      // (b,h,s,r), r fastest
    const int r = g & 31;
    const int s = (g >> 5) & 15;
    const int h = (g >> 9) & 7;
    const int b = g >> 12;
    const int f = b * Ssz + s;
    // element [d][c] of this head/row tile: base + d*1024 + c
    const size_t base = (((size_t)f * Dch + h * DP) * 32 + r) * 32;

    // ---- Q as A fragments (M=c, K=d), K as B fragments (K=d, N=e)
    FragB16 qf[2], kf[2];
#pragma unroll
    for (int mt = 0; mt < 2; ++mt) {
        const int c = mt * 16 + li;
        const int h8 = half * 16;
#pragma unroll
        for (int j = 0; j < 8; ++j) {
            qf[mt].us[j]     = qp[base + (size_t)(h8 + j) * 1024 + c];
            qf[mt].us[8 + j] = qp[base + (size_t)(h8 + 8 + j) * 1024 + c];
        }
    }
#pragma unroll
    for (int nt = 0; nt < 2; ++nt) {
        const int e  = nt * 16 + li;
        const int kb = half * 16;
#pragma unroll
        for (int t = 0; t < 16; ++t)
            kf[nt].us[t] = kp[base + (size_t)(kb + t) * 1024 + e];
    }

    const f32x8 zero = (f32x8){0, 0, 0, 0, 0, 0, 0, 0};
    f32x8 sc[2][2];
#pragma unroll
    for (int mt = 0; mt < 2; ++mt)
#pragma unroll
        for (int nt = 0; nt < 2; ++nt)
            sc[mt][nt] = wmma_bf16(qf[mt].v, kf[nt].v, zero);

    // ---- softmax over e (row = fixed c); row lives across the 16 lanes of
    // one half (lane bits 0..3), both nt accumulators.  ds_swizzle XOR hops.
    const float scale = 0.17677669529663687f;  // 1/sqrt(32)
    float* attn_g = attn_out + (size_t)g * 1024;
#pragma unroll
    for (int mt = 0; mt < 2; ++mt) {
#pragma unroll
        for (int v = 0; v < 8; ++v) {
            float a0 = sc[mt][0][v] * scale;
            float a1 = sc[mt][1][v] * scale;
            float mx = fmaxf(a0, a1);
            mx = fmaxf(mx, swz_xor<0x041F>(mx));   // xor 1
            mx = fmaxf(mx, swz_xor<0x081F>(mx));   // xor 2
            mx = fmaxf(mx, swz_xor<0x101F>(mx));   // xor 4
            mx = fmaxf(mx, swz_xor<0x201F>(mx));   // xor 8
            float e0 = __expf(a0 - mx), e1 = __expf(a1 - mx);
            float sm = e0 + e1;
            sm += swz_xor<0x041F>(sm);
            sm += swz_xor<0x081F>(sm);
            sm += swz_xor<0x101F>(sm);
            sm += swz_xor<0x201F>(sm);
            const float inv = 1.0f / sm;
            const float p0 = e0 * inv, p1 = e1 * inv;
            const int c = mt * 16 + half * 8 + v;
            __builtin_nontemporal_store(p0, attn_g + c * 32 + li);
            __builtin_nontemporal_store(p1, attn_g + c * 32 + 16 + li);
            As[wave][c * 32 + li]      = f2bf(p0);
            As[wave][c * 32 + 16 + li] = f2bf(p1);
        }
    }
    __syncthreads();

    // ---- out = attn (32x32) x V (32x32); A from LDS, B contiguous from vp
    FragB16 vf[2];
#pragma unroll
    for (int nt = 0; nt < 2; ++nt) {
        const int d = nt * 16 + li;
        const size_t a = base + (size_t)d * 1024 + half * 16;
        vf[nt].q[0] = *(const uint4*)(vp + a);
        vf[nt].q[1] = *(const uint4*)(vp + a + 8);
    }
    f32x8 ov[2][2];
#pragma unroll
    for (int mt = 0; mt < 2; ++mt) {
        FragB16 af;
        const unsigned short* row = &As[wave][(mt * 16 + li) * 32 + half * 16];
        af.q[0] = *(const uint4*)(row);
        af.q[1] = *(const uint4*)(row + 8);
#pragma unroll
        for (int nt = 0; nt < 2; ++nt)
            ov[mt][nt] = wmma_bf16(af.v, vf[nt].v, zero);
    }

    // ---- store out6[c,d] -> concat[f, h*32+d, r, c] (bf16)
    const size_t cb = ((size_t)f * Dch + h * DP) * 1024 + (size_t)r * 32;
#pragma unroll
    for (int mt = 0; mt < 2; ++mt)
#pragma unroll
        for (int nt = 0; nt < 2; ++nt)
#pragma unroll
            for (int v = 0; v < 8; ++v) {
                const int c = mt * 16 + half * 8 + v;
                const int d = nt * 16 + li;
                concat[cb + (size_t)d * 1024 + c] = f2bf(ov[mt][nt][v]);
            }
}

// ---------------------------------------------------------------------------
// Launch
// ---------------------------------------------------------------------------
extern "C" void kernel_launch(void* const* d_in, const int* in_sizes, int n_in,
                              void* d_out, int out_size, void* d_ws, size_t ws_size,
                              hipStream_t stream) {
    const float* q    = (const float*)d_in[0];
    const float* k    = (const float*)d_in[1];
    const float* v    = (const float*)d_in[2];
    const float* wq_w = (const float*)d_in[3];
    const float* wq_b = (const float*)d_in[4];
    const float* wk_w = (const float*)d_in[5];
    const float* wk_b = (const float*)d_in[6];
    const float* wv_w = (const float*)d_in[7];
    const float* wv_b = (const float*)d_in[8];
    const float* wo_w = (const float*)d_in[9];
    const float* wo_b = (const float*)d_in[10];

    char* ws = (char*)d_ws;
    unsigned short* W2q = (unsigned short*)ws;            ws += WELEM * 2;
    unsigned short* W2k = (unsigned short*)ws;            ws += WELEM * 2;
    unsigned short* W2v = (unsigned short*)ws;            ws += WELEM * 2;
    unsigned short* W2o = (unsigned short*)ws;            ws += WELEM * 2;
    unsigned short* qp  = (unsigned short*)ws;            ws += AELEM * 2;
    unsigned short* kp  = (unsigned short*)ws;            ws += AELEM * 2;
    unsigned short* vp  = (unsigned short*)ws;            ws += AELEM * 2;
    unsigned short* cc  = (unsigned short*)ws;            ws += AELEM * 2;
    unsigned short* xb  = (unsigned short*)ws;            ws += AELEM * 2;

    float* out  = (float*)d_out;
    float* attn = out + AELEM;

    const int pgrid = (int)((WELEM + 255) / 256);
    pack_weights<<<pgrid, 256, 0, stream>>>(wq_w, W2q);
    pack_weights<<<pgrid, 256, 0, stream>>>(wk_w, W2k);
    pack_weights<<<pgrid, 256, 0, stream>>>(wv_w, W2v);
    pack_weights<<<pgrid, 256, 0, stream>>>(wo_w, W2o);

    const int cvtg = (int)(AELEM / (256 * 8));
    dim3 cgrid(16, FRAMES);  // 4 pos-blocks x 4 chan-blocks, per frame

    cvt_bf16<<<cvtg, 256, 0, stream>>>(q, xb);
    conv3x3_wmma<false><<<cgrid, 256, 0, stream>>>(xb, W2q, wq_b, qp);
    cvt_bf16<<<cvtg, 256, 0, stream>>>(k, xb);
    conv3x3_wmma<false><<<cgrid, 256, 0, stream>>>(xb, W2k, wk_b, kp);
    cvt_bf16<<<cvtg, 256, 0, stream>>>(v, xb);
    conv3x3_wmma<false><<<cgrid, 256, 0, stream>>>(xb, W2v, wv_b, vp);

    attn_wmma<<<(Bsz * Hh * Ssz * 32) / 8, 256, 0, stream>>>(qp, kp, vp, attn, cc);

    conv3x3_wmma<true><<<cgrid, 256, 0, stream>>>(cc, W2o, wo_b, out);
}
